// TransformerBlock_57492432224956
// MI455X (gfx1250) — matrix operand order repaired and checked by reading, not compiled, not run on hardware
//
#include <hip/hip_runtime.h>

// ---------- types ----------
typedef __attribute__((ext_vector_type(16))) __bf16 v16bf;
typedef __attribute__((ext_vector_type(8)))  __bf16 v8bf;
typedef __attribute__((ext_vector_type(8)))  float  v8f;

union BF16Frag { v16bf v; v8bf h[2]; };

__device__ __forceinline__ v8bf zero8bf() {
    v8bf z;
#pragma unroll
    for (int e = 0; e < 8; ++e) z[e] = (__bf16)0.0f;
    return z;
}

// async 16-byte global -> LDS copy (CDNA5, tracked by ASYNCcnt)
__device__ __forceinline__ void async_copy16(unsigned lds_off, const __bf16* g) {
    asm volatile("global_load_async_to_lds_b128 %0, %1, off"
                 :: "v"(lds_off), "v"((unsigned long long)(size_t)g)
                 : "memory");
}

// ---------- problem constants ----------
#define SEQ   2048
#define DIM   2048
#define NH    16
#define HD    128
#define NKV   16
#define FFD   8192
#define QKVN  6144      // (NH + 2*NKV) * HD
#define WIN   256
#define EPSV  1e-5f
#define SCL   0.08838834764831845f   // 1/sqrt(128)

// ---------- workspace layout (bytes) ----------
#define OFF_WQKV 0UL
#define OFF_WO   25165824UL
#define OFF_W1   33554432UL
#define OFF_W3   67108864UL
#define OFF_W2   100663296UL
#define OFF_NX   134217728UL
#define OFF_QKV  142606336UL
#define OFF_Q    192937984UL
#define OFF_K    201326592UL
#define OFF_VT   209715200UL
#define OFF_Y    218103808UL
#define OFF_H    226492416UL
#define OFF_NH   243269632UL
#define OFF_G1   251658240UL
#define OFF_G3   285212672UL
#define OFF_FF   318767104UL

// ======================================================================
// fp32 -> bf16 convert (weights)
// ======================================================================
__global__ __launch_bounds__(256) void k_f32_to_bf16(const float* __restrict__ in,
                                                     __bf16* __restrict__ out, int n) {
    for (int i = blockIdx.x * 256 + threadIdx.x; i < n; i += gridDim.x * 256)
        out[i] = (__bf16)in[i];
}

// ======================================================================
// RMSNorm over rows of length DIM -> bf16
// ======================================================================
__global__ __launch_bounds__(256) void k_rmsnorm_bf16(const float* __restrict__ x,
                                                      const float* __restrict__ w,
                                                      __bf16* __restrict__ out) {
    __shared__ float red[256];
    const int row = blockIdx.x, t = threadIdx.x;
    const float* xr = x + (size_t)row * DIM;
    float ss = 0.f;
    for (int c = t; c < DIM; c += 256) { float v = xr[c]; ss += v * v; }
    red[t] = ss; __syncthreads();
    for (int off = 128; off > 0; off >>= 1) {
        if (t < off) red[t] += red[t + off];
        __syncthreads();
    }
    const float rs = rsqrtf(red[0] / (float)DIM + EPSV);
    for (int c = t; c < DIM; c += 256)
        out[(size_t)row * DIM + c] = (__bf16)(xr[c] * rs * w[c]);
}

// ======================================================================
// bf16 WMMA GEMM:  C[M,N] = A[M,K] * B[N,K]^T   (B row-major [N][K])
// Block tile 64(M) x 256(N), K-step 64, double-buffered ASYNC LDS staging.
// 8 waves, each 32(M) x 64(N): 16 WMMAs / wave / stage.
// EPI: 0 = store f32, 1 = store f32 + resid, 2 = store bf16
// ======================================================================
template <int EPI>
__global__ __launch_bounds__(256) void k_gemm_bf16(const __bf16* __restrict__ A,
                                                   const __bf16* __restrict__ B,
                                                   const float* __restrict__ resid,
                                                   void* __restrict__ outp,
                                                   int M, int N, int K) {
    __shared__ __align__(16) __bf16 As[2][64][64];    // 16 KB
    __shared__ __align__(16) __bf16 Bs[2][256][64];   // 64 KB
    const int t = threadIdx.x;
    const int lane = t & 31, wid = t >> 5;
    const int wm = wid >> 2, wn = wid & 3;            // 2 x 4 wave grid
    const int half = lane >> 4, r16 = lane & 15;
    const int m0 = blockIdx.y * 64, n0 = blockIdx.x * 256;

    v8f acc[2][4];
#pragma unroll
    for (int fa = 0; fa < 2; ++fa)
#pragma unroll
        for (int fb = 0; fb < 4; ++fb)
#pragma unroll
            for (int e = 0; e < 8; ++e) acc[fa][fb][e] = 0.f;

    // issue one stage of async copies: 2 (A) + 8 (B) instructions per thread
    auto issue = [&](int s, int bufI) {
        const int k0 = s * 64;
#pragma unroll
        for (int j = 0; j < 2; ++j) {
            const int c = t + j * 256;            // 512 chunks of 8 bf16
            const int row = c >> 3, col8 = (c & 7) * 8;
            async_copy16((unsigned)(size_t)&As[bufI][row][col8],
                         &A[(size_t)(m0 + row) * K + k0 + col8]);
        }
#pragma unroll
        for (int j = 0; j < 8; ++j) {
            const int c = t + j * 256;            // 2048 chunks of 8 bf16
            const int row = c >> 3, col8 = (c & 7) * 8;
            async_copy16((unsigned)(size_t)&Bs[bufI][row][col8],
                         &B[(size_t)(n0 + row) * K + k0 + col8]);
        }
    };

    const int nstages = K >> 6;
    issue(0, 0);
    for (int s = 0; s < nstages; ++s) {
        const int buf = s & 1;
        if (s + 1 < nstages) {
            issue(s + 1, buf ^ 1);
            // previous stage's 10 copies retire before the 10 just issued
            asm volatile("s_wait_asynccnt 10" ::: "memory");
        } else {
            asm volatile("s_wait_asynccnt 0" ::: "memory");
        }
        __syncthreads();

#pragma unroll
        for (int ks = 0; ks < 64; ks += 32) {
            BF16Frag af[2], bfr[4];
#pragma unroll
            for (int fa = 0; fa < 2; ++fa) {
                const int r = wm * 32 + fa * 16 + r16;
                af[fa].h[0] = *(const v8bf*)&As[buf][r][ks + 8 * half];
                af[fa].h[1] = *(const v8bf*)&As[buf][r][ks + 16 + 8 * half];
            }
#pragma unroll
            for (int fb = 0; fb < 4; ++fb) {
                const int r = wn * 64 + fb * 16 + r16;
                bfr[fb].h[0] = *(const v8bf*)&Bs[buf][r][ks + 8 * half];
                bfr[fb].h[1] = *(const v8bf*)&Bs[buf][r][ks + 16 + 8 * half];
            }
#pragma unroll
            for (int fa = 0; fa < 2; ++fa)
#pragma unroll
                for (int fb = 0; fb < 4; ++fb)
                    acc[fa][fb] = __builtin_amdgcn_wmma_f32_16x16x32_bf16(
                        false, af[fa].v, false, bfr[fb].v, (short)0,
                        acc[fa][fb], false, false);
        }
        __syncthreads();
    }

#pragma unroll
    for (int fa = 0; fa < 2; ++fa)
#pragma unroll
        for (int fb = 0; fb < 4; ++fb)
#pragma unroll
            for (int i = 0; i < 8; ++i) {
                const int row = m0 + wm * 32 + fa * 16 + i + 8 * half;
                const int col = n0 + wn * 64 + fb * 16 + r16;
                const size_t idx = (size_t)row * N + col;
                const float v = acc[fa][fb][i];
                if (EPI == 0) ((float*)outp)[idx] = v;
                else if (EPI == 1) ((float*)outp)[idx] = v + resid[idx];
                else ((__bf16*)outp)[idx] = (__bf16)v;
            }
}

// ======================================================================
// Per-head q/k RMSNorm + rotary; v transposed to [H][HD][S].
// grid (48, S), block 128. head 0..15 -> q, 16..31 -> k, 32..47 -> v
// ======================================================================
__global__ __launch_bounds__(128) void k_qkrope(const float* __restrict__ qkv,
                                                const float* __restrict__ fc,
                                                const float* __restrict__ qw,
                                                const float* __restrict__ kw,
                                                __bf16* __restrict__ qo,
                                                __bf16* __restrict__ ko,
                                                __bf16* __restrict__ vt) {
    const int hidx = blockIdx.x, s = blockIdx.y, t = threadIdx.x;
    const float v = qkv[(size_t)s * QKVN + hidx * HD + t];
    __shared__ float red[128];
    __shared__ float nb[128];
    red[t] = v * v; __syncthreads();
    for (int off = 64; off > 0; off >>= 1) {
        if (t < off) red[t] += red[t + off];
        __syncthreads();
    }
    const float rs = rsqrtf(red[0] / (float)HD + EPSV);
    if (hidx < 32) {
        const float* w = (hidx < 16) ? qw : kw;
        nb[t] = v * rs * w[t];
        __syncthreads();
        const int j = t >> 1;
        const float fr = fc[((size_t)s * (HD / 2) + j) * 2 + 0];
        const float fi = fc[((size_t)s * (HD / 2) + j) * 2 + 1];
        const float xr = nb[j], xi = nb[64 + j];
        const float o = (t & 1) ? (xi * fr + xr * fi) : (xr * fr - xi * fi);
        if (hidx < 16) qo[((size_t)hidx * SEQ + s) * HD + t] = (__bf16)o;
        else ko[((size_t)(hidx - 16) * SEQ + s) * HD + t] = (__bf16)o;
    } else {
        vt[((size_t)(hidx - 32) * HD + t) * SEQ + s] = (__bf16)v;
    }
}

// ======================================================================
// Sliding-window causal flash attention, 1 wave / (16-row q-tile, head).
// q,k: [H][S][HD] bf16; vt: [H][HD][S] bf16; y: [S][H*HD] bf16
// ======================================================================
__device__ __forceinline__ float rmax16(float v) {
    v = fmaxf(v, __shfl_xor(v, 1, 16));
    v = fmaxf(v, __shfl_xor(v, 2, 16));
    v = fmaxf(v, __shfl_xor(v, 4, 16));
    v = fmaxf(v, __shfl_xor(v, 8, 16));
    return v;
}
__device__ __forceinline__ float rsum16(float v) {
    v += __shfl_xor(v, 1, 16);
    v += __shfl_xor(v, 2, 16);
    v += __shfl_xor(v, 4, 16);
    v += __shfl_xor(v, 8, 16);
    return v;
}

__global__ __launch_bounds__(32) void k_attn(const __bf16* __restrict__ q,
                                             const __bf16* __restrict__ k,
                                             const __bf16* __restrict__ vt,
                                             __bf16* __restrict__ y) {
    const int i0 = blockIdx.x * 16;
    const int h = blockIdx.y;
    const int lane = threadIdx.x;
    const int half = lane >> 4, r16 = lane & 15;
    __shared__ __align__(16) __bf16 sp[16][16];

    BF16Frag qf[4];
    const __bf16* qrow = q + ((size_t)h * SEQ + i0 + r16) * HD;
#pragma unroll
    for (int kk = 0; kk < 4; ++kk) {
        qf[kk].h[0] = *(const v8bf*)&qrow[kk * 32 + 8 * half];
        qf[kk].h[1] = *(const v8bf*)&qrow[kk * 32 + 16 + 8 * half];
    }

    v8f o[8];
    float m[8], l[8];
#pragma unroll
    for (int tt = 0; tt < 8; ++tt) {
#pragma unroll
        for (int e = 0; e < 8; ++e) o[tt][e] = 0.f;
        m[tt] = -3.0e38f; l[tt] = 0.f;
    }

    int lomin = i0 - (WIN - 1); if (lomin < 0) lomin = 0;
    const int kb_lo = lomin >> 4, kb_hi = (i0 + 15) >> 4;

    for (int kb = kb_lo; kb <= kb_hi; ++kb) {
        v8f sc;
#pragma unroll
        for (int e = 0; e < 8; ++e) sc[e] = 0.f;
        const __bf16* krow = k + ((size_t)h * SEQ + kb * 16 + r16) * HD;
#pragma unroll
        for (int kk = 0; kk < 4; ++kk) {
            BF16Frag bk;
            bk.h[0] = *(const v8bf*)&krow[kk * 32 + 8 * half];
            bk.h[1] = *(const v8bf*)&krow[kk * 32 + 16 + 8 * half];
            sc = __builtin_amdgcn_wmma_f32_16x16x32_bf16(
                false, qf[kk].v, false, bk.v, (short)0, sc, false, false);
        }

        const int kcol = kb * 16 + r16;
        float fsc[8];
#pragma unroll
        for (int i = 0; i < 8; ++i) {
            const int qr = i0 + i + 8 * half;
            const bool ok = (kcol <= qr) && (qr - kcol < WIN);
            const float sv = ok ? sc[i] * SCL : -3.0e38f;
            const float bm = rmax16(sv);
            const float mn = fmaxf(m[i], bm);
            const float f = __expf(m[i] - mn);
            const float p = ok ? __expf(sv - mn) : 0.f;
            l[i] = l[i] * f + rsum16(p);
            m[i] = mn;
            fsc[i] = f;
            sp[i + 8 * half][r16] = (__bf16)p;
        }
#pragma unroll
        for (int tt = 0; tt < 8; ++tt)
#pragma unroll
            for (int i = 0; i < 8; ++i) o[tt][i] *= fsc[i];
        __syncthreads();

        BF16Frag pa;
        pa.h[0] = *(const v8bf*)&sp[r16][8 * half];
        pa.h[1] = zero8bf();
#pragma unroll
        for (int tt = 0; tt < 8; ++tt) {
            BF16Frag vb;
            {
                const __bf16* vr =
                    vt + ((size_t)h * HD + tt * 16 + r16) * SEQ + kb * 16;
                vb.h[0] = *(const v8bf*)&vr[8 * half];
                vb.h[1] = zero8bf();
            }
            o[tt] = __builtin_amdgcn_wmma_f32_16x16x32_bf16(
                false, pa.v, false, vb.v, (short)0, o[tt], false, false);
        }
        __syncthreads();
    }

#pragma unroll
    for (int tt = 0; tt < 8; ++tt)
#pragma unroll
        for (int i = 0; i < 8; ++i) {
            const int row = i0 + i + 8 * half;
            const float val = o[tt][i] / l[i];
            y[(size_t)row * DIM + h * HD + tt * 16 + r16] = (__bf16)val;
        }
}

// ======================================================================
// silu(g1) * g3 -> bf16
// ======================================================================
__global__ __launch_bounds__(256) void k_silumul(const __bf16* __restrict__ g1,
                                                 const __bf16* __restrict__ g3,
                                                 __bf16* __restrict__ out, int n) {
    for (int i = blockIdx.x * 256 + threadIdx.x; i < n; i += gridDim.x * 256) {
        const float a = (float)g1[i];
        const float b = (float)g3[i];
        const float s = a / (1.f + __expf(-a));
        out[i] = (__bf16)(s * b);
    }
}

// ======================================================================
// launch
// ======================================================================
extern "C" void kernel_launch(void* const* d_in, const int* in_sizes, int n_in,
                              void* d_out, int out_size, void* d_ws, size_t ws_size,
                              hipStream_t stream) {
    (void)in_sizes; (void)n_in; (void)out_size; (void)ws_size;
    const float* x   = (const float*)d_in[0];
    const float* fc  = (const float*)d_in[2];
    const float* wqkv= (const float*)d_in[4];
    const float* wo  = (const float*)d_in[5];
    const float* qnw = (const float*)d_in[6];
    const float* knw = (const float*)d_in[7];
    const float* anw = (const float*)d_in[8];
    const float* fnw = (const float*)d_in[9];
    const float* w1  = (const float*)d_in[10];
    const float* w3  = (const float*)d_in[11];
    const float* w2  = (const float*)d_in[12];

    char* ws = (char*)d_ws;
    __bf16* wqkv_b = (__bf16*)(ws + OFF_WQKV);
    __bf16* wo_b   = (__bf16*)(ws + OFF_WO);
    __bf16* w1_b   = (__bf16*)(ws + OFF_W1);
    __bf16* w3_b   = (__bf16*)(ws + OFF_W3);
    __bf16* w2_b   = (__bf16*)(ws + OFF_W2);
    __bf16* nx_b   = (__bf16*)(ws + OFF_NX);
    float*  qkv_f  = (float*) (ws + OFF_QKV);
    __bf16* q_b    = (__bf16*)(ws + OFF_Q);
    __bf16* k_b    = (__bf16*)(ws + OFF_K);
    __bf16* vt_b   = (__bf16*)(ws + OFF_VT);
    __bf16* y_b    = (__bf16*)(ws + OFF_Y);
    float*  h_f    = (float*) (ws + OFF_H);
    __bf16* nh_b   = (__bf16*)(ws + OFF_NH);
    __bf16* g1_b   = (__bf16*)(ws + OFF_G1);
    __bf16* g3_b   = (__bf16*)(ws + OFF_G3);
    __bf16* ff_b   = (__bf16*)(ws + OFF_FF);

    k_f32_to_bf16<<<4096, 256, 0, stream>>>(wqkv, wqkv_b, QKVN * DIM);
    k_f32_to_bf16<<<4096, 256, 0, stream>>>(wo,   wo_b,   DIM * DIM);
    k_f32_to_bf16<<<4096, 256, 0, stream>>>(w1,   w1_b,   FFD * DIM);
    k_f32_to_bf16<<<4096, 256, 0, stream>>>(w3,   w3_b,   FFD * DIM);
    k_f32_to_bf16<<<4096, 256, 0, stream>>>(w2,   w2_b,   DIM * FFD);

    k_rmsnorm_bf16<<<SEQ, 256, 0, stream>>>(x, anw, nx_b);

    k_gemm_bf16<0><<<dim3(QKVN / 256, SEQ / 64), 256, 0, stream>>>(
        nx_b, wqkv_b, nullptr, qkv_f, SEQ, QKVN, DIM);

    k_qkrope<<<dim3(48, SEQ), 128, 0, stream>>>(qkv_f, fc, qnw, knw, q_b, k_b, vt_b);

    k_attn<<<dim3(SEQ / 16, NH), 32, 0, stream>>>(q_b, k_b, vt_b, y_b);

    k_gemm_bf16<1><<<dim3(DIM / 256, SEQ / 64), 256, 0, stream>>>(
        y_b, wo_b, x, h_f, SEQ, DIM, DIM);

    k_rmsnorm_bf16<<<SEQ, 256, 0, stream>>>(h_f, fnw, nh_b);

    k_gemm_bf16<2><<<dim3(FFD / 256, SEQ / 64), 256, 0, stream>>>(
        nh_b, w1_b, nullptr, g1_b, SEQ, FFD, DIM);
    k_gemm_bf16<2><<<dim3(FFD / 256, SEQ / 64), 256, 0, stream>>>(
        nh_b, w3_b, nullptr, g3_b, SEQ, FFD, DIM);

    k_silumul<<<4096, 256, 0, stream>>>(g1_b, g3_b, ff_b, SEQ * FFD);

    k_gemm_bf16<1><<<dim3(DIM / 256, SEQ / 64), 256, 0, stream>>>(
        ff_b, w2_b, h_f, (float*)d_out, SEQ, DIM, FFD);
}
